// CMRM_46703474377216
// MI455X (gfx1250) — compile-verified, matrix operand order checked by reading, not run on hardware
//
#include <hip/hip_runtime.h>
#include <hip/hip_bf16.h>
#include <math.h>

#define DIMX   4096
#define NB     8
#define NSEQ   2048
#define NTOK   (NB*NSEQ)        // 16384
#define NSLOTS 7
#define NHEADS 8
#define HDIMX  512
#define RROWS  56               // h*7+q rows used
#define MROWS  64               // padded M for WMMA
#define SCALEQ 0.04419417382415922f   // 1/sqrt(512)

typedef __attribute__((ext_vector_type(16))) __bf16 v16bf;
typedef __attribute__((ext_vector_type(8)))  __bf16 v8bf;
typedef __attribute__((ext_vector_type(8)))  float  v8f;

// ---------- bf16 helpers (bit-level, avoid relying on __bf16 arithmetic) ----------
__device__ __forceinline__ unsigned short f2bf(float f) {
    unsigned int u = __float_as_uint(f);
    u += 0x7FFFu + ((u >> 16) & 1u);        // round-to-nearest-even
    return (unsigned short)(u >> 16);
}
__device__ __forceinline__ float bf2f(unsigned short h) {
    return __uint_as_float(((unsigned int)h) << 16);
}

// Load one 16-bf16 WMMA fragment for lane (hi selects K-subblocks).
// Per ISA layout: elements 0..7 <- K = off..off+7, elements 8..15 <- K = off+16..off+23
__device__ __forceinline__ v16bf load_frag(const unsigned short* __restrict__ p, int off) {
    v8bf a = *(const v8bf*)(p + off);
    v8bf b = *(const v8bf*)(p + off + 16);
    v16bf r;
#pragma unroll
    for (int i = 0; i < 8; ++i) { r[i] = a[i]; r[i + 8] = b[i]; }
    return r;
}

#define WMMA_BF16(A, B, C) \
    __builtin_amdgcn_wmma_f32_16x16x32_bf16(false, (A), false, (B), (short)0, (C), false, false)

// ================== Kernel A1: q = clip(slots) @ Wq^T + bq  (7 x 4096) ==================
__global__ void qproj_kernel(const float* __restrict__ slots, const float* __restrict__ ipw,
                             const float* __restrict__ ipb, float* __restrict__ q_ws) {
    int d = blockIdx.x * 256 + threadIdx.x;           // 16 blocks
    float acc[NSLOTS];
#pragma unroll
    for (int qi = 0; qi < NSLOTS; ++qi) acc[qi] = 0.f;
    const float* wrow = ipw + (size_t)d * DIMX;       // Wq row d
    for (int e = 0; e < DIMX; ++e) {
        float w = wrow[e];
#pragma unroll
        for (int qi = 0; qi < NSLOTS; ++qi) {
            float sv = slots[qi * DIMX + e];
            sv = fminf(10.f, fmaxf(-10.f, sv));
            acc[qi] += sv * w;
        }
    }
#pragma unroll
    for (int qi = 0; qi < NSLOTS; ++qi) q_ws[qi * DIMX + d] = acc[qi] + ipb[d];
}

// ====== Kernel A2: qk[r=h*7+q][e] = scale * sum_dh q[q][h*512+dh]*Wk[h*512+dh][e]  (bf16) ======
__global__ void qkmat_kernel(const float* __restrict__ q_ws, const float* __restrict__ ipw,
                             unsigned short* __restrict__ qk_ws) {
    int h = blockIdx.x >> 4;                           // 128 blocks = 8h x 16 e-tiles
    int e = (blockIdx.x & 15) * 256 + threadIdx.x;
    float acc[NSLOTS];
#pragma unroll
    for (int qi = 0; qi < NSLOTS; ++qi) acc[qi] = 0.f;
    for (int dh = 0; dh < HDIMX; ++dh) {
        int drow = h * HDIMX + dh;
        float w = ipw[((size_t)(DIMX + drow)) * DIMX + e];   // Wk row
#pragma unroll
        for (int qi = 0; qi < NSLOTS; ++qi) acc[qi] += q_ws[qi * DIMX + drow] * w;
    }
#pragma unroll
    for (int qi = 0; qi < NSLOTS; ++qi)
        qk_ws[(size_t)(h * NSLOTS + qi) * DIMX + e] = f2bf(acc[qi] * SCALEQ);
}

// ====== Kernel A3: qb[r] = scale * sum_dh q*bk ; zero pad rows 56..63 of qk ======
__global__ void qbias_kernel(const float* __restrict__ q_ws, const float* __restrict__ ipb,
                             float* __restrict__ qb, unsigned short* __restrict__ qk_ws) {
    int t = threadIdx.x;
    if (t < MROWS) {
        float s = 0.f;
        if (t < RROWS) {
            int h = t / NSLOTS, qi = t % NSLOTS;
            for (int dh = 0; dh < HDIMX; ++dh) {
                int drow = h * HDIMX + dh;
                s += q_ws[qi * DIMX + drow] * ipb[DIMX + drow];
            }
            s *= SCALEQ;
        }
        qb[t] = s;
    }
    for (int i = t; i < (MROWS - RROWS) * DIMX; i += 256)
        qk_ws[RROWS * DIMX + i] = 0;
}

// ================== Kernel B: LayerNorm -> xn (bf16), one token per block ==================
__global__ void ln_kernel(const float* __restrict__ x, const float* __restrict__ g,
                          const float* __restrict__ be, unsigned short* __restrict__ xn) {
    __shared__ float red[256];
    int t = threadIdx.x;
    size_t tok = blockIdx.x;
    const float* row = x + tok * DIMX;
    float v[16];
    float s = 0.f;
#pragma unroll
    for (int j = 0; j < 4; ++j) {
        float4 f = *(const float4*)(row + t * 16 + j * 4);
        v[4 * j + 0] = f.x; v[4 * j + 1] = f.y; v[4 * j + 2] = f.z; v[4 * j + 3] = f.w;
        s += f.x + f.y + f.z + f.w;
    }
    red[t] = s; __syncthreads();
    for (int o = 128; o; o >>= 1) { if (t < o) red[t] += red[t + o]; __syncthreads(); }
    float mu = red[0] * (1.f / DIMX); __syncthreads();
    float vs = 0.f;
#pragma unroll
    for (int j = 0; j < 16; ++j) { float d = v[j] - mu; vs += d * d; }
    red[t] = vs; __syncthreads();
    for (int o = 128; o; o >>= 1) { if (t < o) red[t] += red[t + o]; __syncthreads(); }
    float rstd = rsqrtf(red[0] * (1.f / DIMX) + 1e-5f);
    unsigned int pk[8];
#pragma unroll
    for (int j = 0; j < 8; ++j) {
        int d0 = t * 16 + 2 * j;
        float y0 = (v[2 * j]     - mu) * rstd * g[d0]     + be[d0];
        float y1 = (v[2 * j + 1] - mu) * rstd * g[d0 + 1] + be[d0 + 1];
        pk[j] = (unsigned int)f2bf(y0) | ((unsigned int)f2bf(y1) << 16);
    }
    uint4* dst = (uint4*)(xn + tok * DIMX + t * 16);
    dst[0] = make_uint4(pk[0], pk[1], pk[2], pk[3]);
    dst[1] = make_uint4(pk[4], pk[5], pk[6], pk[7]);
}

// ====== Kernel C: scores GEMM via WMMA bf16, software-pipelined. ======
// Each wave: 64(M) x 32(tokens, two 16-wide B tiles), K = 4096.
// Per k-step: 6 fragment loads (12 x b128) feed 8 WMMAs; next-k fragments are
// prefetched into a second register set so loads overlap the matrix ops.
__global__ void __launch_bounds__(128) scores_kernel(const unsigned short* __restrict__ xn,
                                                     const unsigned short* __restrict__ qk,
                                                     const float* __restrict__ qb,
                                                     float* __restrict__ scores) {
    int wave = threadIdx.x >> 5;
    int lane = threadIdx.x & 31;
    int hi = lane >> 4, lo = lane & 15;
    int tok0 = (blockIdx.x * 4 + wave) * 32;           // 128 blocks x 4 waves x 32 tokens
    const unsigned short* b0r = xn + (size_t)(tok0 + lo) * DIMX;       // B = xn^T : K contiguous
    const unsigned short* b1r = xn + (size_t)(tok0 + 16 + lo) * DIMX;
    const unsigned short* a0r = qk + (size_t)(lo) * DIMX;
    const unsigned short* a1r = qk + (size_t)(16 + lo) * DIMX;
    const unsigned short* a2r = qk + (size_t)(32 + lo) * DIMX;
    const unsigned short* a3r = qk + (size_t)(48 + lo) * DIMX;

    v8f c00 = {}, c01 = {}, c10 = {}, c11 = {};
    v8f c20 = {}, c21 = {}, c30 = {}, c31 = {};

    int off = hi * 8;
    v16bf B0 = load_frag(b0r, off), B1 = load_frag(b1r, off);
    v16bf A0 = load_frag(a0r, off), A1 = load_frag(a1r, off);
    v16bf A2 = load_frag(a2r, off), A3 = load_frag(a3r, off);

    for (int k0 = 0; k0 < DIMX - 32; k0 += 32) {
        int noff = k0 + 32 + hi * 8;
        v16bf nB0 = load_frag(b0r, noff), nB1 = load_frag(b1r, noff);
        v16bf nA0 = load_frag(a0r, noff), nA1 = load_frag(a1r, noff);
        v16bf nA2 = load_frag(a2r, noff), nA3 = load_frag(a3r, noff);

        c00 = WMMA_BF16(A0, B0, c00);  c01 = WMMA_BF16(A0, B1, c01);
        c10 = WMMA_BF16(A1, B0, c10);  c11 = WMMA_BF16(A1, B1, c11);
        c20 = WMMA_BF16(A2, B0, c20);  c21 = WMMA_BF16(A2, B1, c21);
        c30 = WMMA_BF16(A3, B0, c30);  c31 = WMMA_BF16(A3, B1, c31);

        B0 = nB0; B1 = nB1; A0 = nA0; A1 = nA1; A2 = nA2; A3 = nA3;
    }
    // peeled final k-step
    c00 = WMMA_BF16(A0, B0, c00);  c01 = WMMA_BF16(A0, B1, c01);
    c10 = WMMA_BF16(A1, B0, c10);  c11 = WMMA_BF16(A1, B1, c11);
    c20 = WMMA_BF16(A2, B0, c20);  c21 = WMMA_BF16(A2, B1, c21);
    c30 = WMMA_BF16(A3, B0, c30);  c31 = WMMA_BF16(A3, B1, c31);

    // D layout: VGPR i, lane l -> M = i + 8*(l>>4), N = l&15
#pragma unroll
    for (int tile = 0; tile < 2; ++tile) {
        int tokenc = tok0 + tile * 16 + lo;
        int b = tokenc >> 11;                  // NSEQ = 2048
        int n = tokenc & (NSEQ - 1);
        float* sb = scores + (size_t)b * MROWS * NSEQ + n;
        const v8f* cc0 = tile ? &c01 : &c00;
        const v8f* cc1 = tile ? &c11 : &c10;
        const v8f* cc2 = tile ? &c21 : &c20;
        const v8f* cc3 = tile ? &c31 : &c30;
#pragma unroll
        for (int i = 0; i < 8; ++i) {
            int m0 = i + hi * 8;
            sb[(size_t)(m0)      * NSEQ] = (*cc0)[i] + qb[m0];
            sb[(size_t)(m0 + 16) * NSEQ] = (*cc1)[i] + qb[m0 + 16];
            sb[(size_t)(m0 + 32) * NSEQ] = (*cc2)[i] + qb[m0 + 32];
            sb[(size_t)(m0 + 48) * NSEQ] = (*cc3)[i] + qb[m0 + 48];
        }
    }
}

// ====== Kernel D: softmax over N, attn_weights (mean over h), A = sum_q attn, S = sum_n A ======
__global__ void softmax_kernel(const float* __restrict__ scores,
                               float* __restrict__ A_ws, float* __restrict__ S_ws,
                               float* __restrict__ attn_out) {
    __shared__ float red[256];
    int t = threadIdx.x;
    int b = blockIdx.x;                                 // 8 blocks
    float wacc[NSLOTS][8];
#pragma unroll
    for (int q = 0; q < NSLOTS; ++q)
#pragma unroll
        for (int j = 0; j < 8; ++j) wacc[q][j] = 0.f;
    for (int h = 0; h < NHEADS; ++h) {
        float aacc[8];
#pragma unroll
        for (int j = 0; j < 8; ++j) aacc[j] = 0.f;
        for (int q = 0; q < NSLOTS; ++q) {
            const float* row = scores + ((size_t)(b * MROWS + h * NSLOTS + q)) * NSEQ;
            float v[8];
            float4 v0 = *(const float4*)(row + t * 8);
            float4 v1 = *(const float4*)(row + t * 8 + 4);
            v[0] = v0.x; v[1] = v0.y; v[2] = v0.z; v[3] = v0.w;
            v[4] = v1.x; v[5] = v1.y; v[6] = v1.z; v[7] = v1.w;
            float m = v[0];
#pragma unroll
            for (int j = 1; j < 8; ++j) m = fmaxf(m, v[j]);
            red[t] = m; __syncthreads();
            for (int o = 128; o; o >>= 1) { if (t < o) red[t] = fmaxf(red[t], red[t + o]); __syncthreads(); }
            float mx = red[0]; __syncthreads();
            float s = 0.f;
#pragma unroll
            for (int j = 0; j < 8; ++j) { v[j] = __expf(v[j] - mx); s += v[j]; }
            red[t] = s; __syncthreads();
            for (int o = 128; o; o >>= 1) { if (t < o) red[t] += red[t + o]; __syncthreads(); }
            float inv = 1.f / red[0]; __syncthreads();
#pragma unroll
            for (int j = 0; j < 8; ++j) { float p = v[j] * inv; aacc[j] += p; wacc[q][j] += p; }
        }
        float ls = 0.f;
#pragma unroll
        for (int j = 0; j < 8; ++j) {
            A_ws[((size_t)(b * NHEADS + h)) * NSEQ + t * 8 + j] = aacc[j];
            ls += aacc[j];
        }
        red[t] = ls; __syncthreads();
        for (int o = 128; o; o >>= 1) { if (t < o) red[t] += red[t + o]; __syncthreads(); }
        if (t == 0) S_ws[b * NHEADS + h] = red[0];
        __syncthreads();
    }
#pragma unroll
    for (int q = 0; q < NSLOTS; ++q)
#pragma unroll
        for (int j = 0; j < 8; ++j)
            attn_out[((size_t)(b * NSLOTS + q)) * NSEQ + t * 8 + j] = wacc[q][j] * (1.f / NHEADS);
}

// ====== Kernel E: y[b,h,e] = sum_n A[b,h,n] * xn[b,n,e] ======
__global__ void ygemm_kernel(const unsigned short* __restrict__ xn, const float* __restrict__ A_ws,
                             float* __restrict__ y_ws) {
    int b = blockIdx.x >> 4;                            // 128 blocks = 8b x 16 e-tiles
    int e = (blockIdx.x & 15) * 256 + threadIdx.x;
    float acc[NHEADS];
#pragma unroll
    for (int h = 0; h < NHEADS; ++h) acc[h] = 0.f;
    const float* Ab = A_ws + (size_t)b * NHEADS * NSEQ;
    const unsigned short* xb = xn + (size_t)b * NSEQ * DIMX + e;
    for (int n = 0; n < NSEQ; ++n) {
        float xv = bf2f(xb[(size_t)n * DIMX]);
#pragma unroll
        for (int h = 0; h < NHEADS; ++h) acc[h] += Ab[h * NSEQ + n] * xv;
    }
#pragma unroll
    for (int h = 0; h < NHEADS; ++h) y_ws[(size_t)(b * NHEADS + h) * DIMX + e] = acc[h];
}

// ====== Kernel F1: sctx[b,d] = sum_e y[b,h(d),e]*Wv[d,e] + S[b,h]*bv[d] ======
__global__ void sctx_kernel(const float* __restrict__ y_ws, const float* __restrict__ ipw,
                            const float* __restrict__ ipb, const float* __restrict__ S_ws,
                            float* __restrict__ sctx) {
    int d = blockIdx.x * 256 + threadIdx.x;             // 16 blocks
    int h = d >> 9;                                     // HDIMX = 512
    float acc[NB];
#pragma unroll
    for (int b = 0; b < NB; ++b) acc[b] = 0.f;
    const float* wrow = ipw + ((size_t)(2 * DIMX + d)) * DIMX;   // Wv row d
    for (int e = 0; e < DIMX; ++e) {
        float w = wrow[e];
#pragma unroll
        for (int b = 0; b < NB; ++b) acc[b] += y_ws[(size_t)(b * NHEADS + h) * DIMX + e] * w;
    }
    float bv = ipb[2 * DIMX + d];
#pragma unroll
    for (int b = 0; b < NB; ++b) sctx[b * DIMX + d] = acc[b] + S_ws[b * NHEADS + h] * bv;
}

// ====== Kernel F2: gvec[b,d] = tanh(alpha)/7 * ( sum_q slots_c + sum_e Wo[d,e]*sctx[b,e] + 7*bo[d] ) ======
__global__ void gvec_kernel(const float* __restrict__ sctx, const float* __restrict__ opw,
                            const float* __restrict__ opb, const float* __restrict__ slots,
                            const float* __restrict__ alpha, float* __restrict__ gvec) {
    int d = blockIdx.x * 256 + threadIdx.x;             // 16 blocks
    float acc[NB];
#pragma unroll
    for (int b = 0; b < NB; ++b) acc[b] = 0.f;
    const float* wrow = opw + (size_t)d * DIMX;
    for (int e = 0; e < DIMX; ++e) {
        float w = wrow[e];
#pragma unroll
        for (int b = 0; b < NB; ++b) acc[b] += sctx[b * DIMX + e] * w;
    }
    float ss = 0.f;
#pragma unroll
    for (int qi = 0; qi < NSLOTS; ++qi) {
        float sv = slots[qi * DIMX + d];
        ss += fminf(10.f, fmaxf(-10.f, sv));
    }
    float gate = tanhf(alpha[0]);
    float bo7 = (float)NSLOTS * opb[d];
#pragma unroll
    for (int b = 0; b < NB; ++b)
        gvec[b * DIMX + d] = gate * (ss + acc[b] + bo7) * (1.f / NSLOTS);
}

// ====== Kernel G: out = x + gvec[b,:] broadcast ======
__global__ void add_kernel(const float* __restrict__ x, const float* __restrict__ gvec,
                           float* __restrict__ out) {
    size_t i4 = (size_t)blockIdx.x * blockDim.x + threadIdx.x;
    size_t i = i4 * 4;
    int d = (int)(i & (DIMX - 1));
    int b = (int)(i >> 23);                             // N*DIMX = 2^23
    float4 xv = *(const float4*)(x + i);
    float4 gv = *(const float4*)(gvec + (size_t)b * DIMX + d);
    float4 o;
    o.x = xv.x + gv.x; o.y = xv.y + gv.y; o.z = xv.z + gv.z; o.w = xv.w + gv.w;
    *(float4*)(out + i) = o;
}

// ================== Workspace layout (bytes) ==================
#define WS_XN   0ull
#define WS_Q    134217728ull                 // xn: 16384*4096*2
#define WS_QK   (WS_Q   + 114688ull)         // q:  7*4096*4
#define WS_QB   (WS_QK  + 524288ull)         // qk: 64*4096*2
#define WS_SC   (WS_QB  + 256ull)            // qb: 64*4
#define WS_A    (WS_SC  + 4194304ull)        // scores: 8*64*2048*4
#define WS_S    (WS_A   + 524288ull)         // A: 8*8*2048*4
#define WS_Y    (WS_S   + 256ull)            // S: 64*4
#define WS_CTX  (WS_Y   + 1048576ull)        // y: 8*8*4096*4
#define WS_G    (WS_CTX + 131072ull)         // sctx: 8*4096*4

extern "C" void kernel_launch(void* const* d_in, const int* in_sizes, int n_in,
                              void* d_out, int out_size, void* d_ws, size_t ws_size,
                              hipStream_t stream) {
    const float* x     = (const float*)d_in[0];
    const float* slots = (const float*)d_in[1];
    const float* ln_g  = (const float*)d_in[2];
    const float* ln_b  = (const float*)d_in[3];
    const float* ipw   = (const float*)d_in[4];   // [3*4096, 4096] rows = [Wq; Wk; Wv]
    const float* ipb   = (const float*)d_in[5];
    const float* opw   = (const float*)d_in[6];
    const float* opb   = (const float*)d_in[7];
    const float* alpha = (const float*)d_in[8];

    float* out_x = (float*)d_out;                         // (8,2048,4096) f32
    float* out_w = (float*)d_out + 67108864;              // (8,7,2048) f32

    char* ws = (char*)d_ws;
    unsigned short* xn    = (unsigned short*)(ws + WS_XN);
    float*          q_ws  = (float*)(ws + WS_Q);
    unsigned short* qk_ws = (unsigned short*)(ws + WS_QK);
    float*          qb_ws = (float*)(ws + WS_QB);
    float*          sc_ws = (float*)(ws + WS_SC);
    float*          A_ws  = (float*)(ws + WS_A);
    float*          S_ws  = (float*)(ws + WS_S);
    float*          y_ws  = (float*)(ws + WS_Y);
    float*          c_ws  = (float*)(ws + WS_CTX);
    float*          g_ws  = (float*)(ws + WS_G);

    qproj_kernel  <<<16,    256, 0, stream>>>(slots, ipw, ipb, q_ws);
    qkmat_kernel  <<<128,   256, 0, stream>>>(q_ws, ipw, qk_ws);
    qbias_kernel  <<<1,     256, 0, stream>>>(q_ws, ipb, qb_ws, qk_ws);
    ln_kernel     <<<16384, 256, 0, stream>>>(x, ln_g, ln_b, xn);
    scores_kernel <<<128,   128, 0, stream>>>(xn, qk_ws, qb_ws, sc_ws);
    softmax_kernel<<<8,     256, 0, stream>>>(sc_ws, A_ws, S_ws, out_w);
    ygemm_kernel  <<<128,   256, 0, stream>>>(xn, A_ws, y_ws);
    sctx_kernel   <<<16,    256, 0, stream>>>(y_ws, ipw, ipb, S_ws, c_ws);
    gvec_kernel   <<<16,    256, 0, stream>>>(c_ws, opw, opb, slots, alpha, g_ws);
    add_kernel    <<<65536, 256, 0, stream>>>(x, g_ws, out_x);
}
